// Attention_11244224381457
// MI455X (gfx1250) — matrix operand order repaired and checked by reading, not compiled, not run on hardware
//
#include <hip/hip_runtime.h>

// ---------------------------------------------------------------------------
// Causal MHA for MI455X (gfx1250, wave32, WMMA bf16 16x16x32, f32 accum)
// B=2, S=2048, D=1024, H=16, A=64
// Async global->LDS staging (ASYNCcnt), double-buffered tiles, prefetch.
// ---------------------------------------------------------------------------

#define B_ 2
#define S_ 2048
#define D_ 1024
#define H_ 16
#define A_ 64

typedef __attribute__((ext_vector_type(16))) __bf16 bf16x16;
typedef __attribute__((ext_vector_type(8)))  float  f32x8;

union Frag16 { bf16x16 v; uint4 q[2]; };
union Pack8  { uint4 q; __bf16 e[8]; };
union Pack4  { uint2 u; __bf16 e[4]; };

static __device__ __forceinline__ f32x8 zero8() {
  f32x8 z = {0.f,0.f,0.f,0.f,0.f,0.f,0.f,0.f};
  return z;
}

static __device__ __forceinline__ f32x8 wmma_bf16(const Frag16& a, const Frag16& b, f32x8 c) {
  return __builtin_amdgcn_wmma_f32_16x16x32_bf16(false, a.v, false, b.v, (short)0, c, false, false);
}

// Async global->LDS 16B copy, tracked by ASYNCcnt. VDST is the LDS byte
// address (low 32 bits of the generic LDS pointer per the aperture mapping).
static __device__ __forceinline__ void async_b128(void* lds_dst, const void* gsrc) {
  asm volatile("global_load_async_to_lds_b128 %0, %1, off"
               :: "v"((unsigned)(unsigned long long)lds_dst), "v"(gsrc)
               : "memory");
}
static __device__ __forceinline__ void wait_async0() {
  asm volatile("s_wait_asynccnt 0" ::: "memory");
}
static __device__ __forceinline__ void wait_ds0() {
  asm volatile("s_wait_dscnt 0" ::: "memory");
}

// ---------------------------------------------------------------------------
// fp32 -> bf16 conversion (4 elements / thread, vectorized)
// ---------------------------------------------------------------------------
__global__ __launch_bounds__(256) void k_cvt(const float* __restrict__ src,
                                             __bf16* __restrict__ dst, int n4) {
  int i = blockIdx.x * blockDim.x + threadIdx.x;
  if (i >= n4) return;
  float4 f = ((const float4*)src)[i];
  Pack4 p;
  p.e[0] = (__bf16)f.x; p.e[1] = (__bf16)f.y;
  p.e[2] = (__bf16)f.z; p.e[3] = (__bf16)f.w;
  ((uint2*)dst)[i] = p.u;
}

// ---------------------------------------------------------------------------
// QKV projection: out[b,h,s,a] = residual[b,s,:] @ W[h,:,a] + bias[h,a]
// grid: (B*S/128, H, 3), block 256 (8 waves). 128x64 tile, K-step 32,
// double-buffered LDS, async A-tile staging.
// ---------------------------------------------------------------------------
__global__ __launch_bounds__(256) void k_qkv(
    const __bf16* __restrict__ xb,                               // [B,S,D]
    const __bf16* __restrict__ wq, const __bf16* __restrict__ wk,
    const __bf16* __restrict__ wv,                               // [H,D,A]
    const float* __restrict__ bq, const float* __restrict__ bk,
    const float* __restrict__ bv,                                // [H,A]
    __bf16* __restrict__ qb, __bf16* __restrict__ kb,
    __bf16* __restrict__ vb)                                     // [B,H,S,A]
{
  __shared__ __bf16 At[2][128 * 40];  // A tile [128][32] bf16, stride 40
  __shared__ __bf16 Bt[2][64 * 40];   // B tile transposed [N=64][K=32]

  const int mt    = blockIdx.x % (S_ / 128);
  const int b     = blockIdx.x / (S_ / 128);
  const int h     = blockIdx.y;
  const int which = blockIdx.z;

  const __bf16* W    = (which == 0) ? wq : (which == 1) ? wk : wv;
  const float*  bias = (which == 0) ? bq : (which == 1) ? bk : bv;
  __bf16*       out  = (which == 0) ? qb : (which == 1) ? kb : vb;

  const int tid  = threadIdx.x;
  const int lane = tid & 31;
  const int w    = tid >> 5;
  const int m0   = mt * 128;

  const __bf16* Ab = xb + ((size_t)b * S_ + m0) * D_;
  const __bf16* Wb = W + (size_t)h * D_ * A_;

  f32x8 acc[4] = { zero8(), zero8(), zero8(), zero8() };

  const int ar  = tid >> 1;        // 0..127 : A-tile row
  const int ah  = tid & 1;         // 16-elem half of the 32-wide row
  const int bn  = tid & 63;        // B-tile column (N)
  const int bk8 = (tid >> 6) * 8;  // B-tile K group
  const int row16 = lane & 15;
  const int half  = lane >> 4;

  // stage helpers -----------------------------------------------------------
  auto stageA = [&](int buf, int kk) {
    const __bf16* src = Ab + (size_t)ar * D_ + kk + ah * 16;
    __bf16* dst = &At[buf][ar * 40 + ah * 16];
    async_b128(dst, src);
    async_b128(dst + 8, src + 8);
  };
  auto stageB = [&](int buf, int kk) {
    #pragma unroll
    for (int j = 0; j < 8; ++j)
      Bt[buf][bn * 40 + bk8 + j] = Wb[(size_t)(kk + bk8 + j) * A_ + bn];
  };

  stageA(0, 0);
  stageB(0, 0);

  for (int it = 0; it < D_ / 32; ++it) {
    const int kk  = it * 32;
    const int cur = it & 1;
    wait_async0();
    __syncthreads();
    if (kk + 32 < D_) {
      stageA(cur ^ 1, kk + 32);
      stageB(cur ^ 1, kk + 32);
      if (kk + 64 < D_) {
        __builtin_prefetch(Ab + (size_t)ar * D_ + kk + 64 + ah * 16);
        if (tid < 32) __builtin_prefetch(Wb + (size_t)(kk + 64 + tid) * A_);
      }
    }

    // A fragment: row = lane%16 of this wave's strip
    Frag16 a;
    {
      const __bf16* p = &At[cur][(w * 16 + row16) * 40 + half * 16];
      a.q[0] = *(const uint4*)(p);        // K = kbase + 0..7
      a.q[1] = *(const uint4*)(p + 8);    // K = kbase + 8..15
    }
    // batch all 4 B fragments, then issue WMMAs back-to-back
    Frag16 bfr[4];
    #pragma unroll
    for (int nt = 0; nt < 4; ++nt) {
      const __bf16* p = &Bt[cur][(nt * 16 + row16) * 40 + half * 16];
      bfr[nt].q[0] = *(const uint4*)(p);
      bfr[nt].q[1] = *(const uint4*)(p + 8);
    }
    #pragma unroll
    for (int nt = 0; nt < 4; ++nt)
      acc[nt] = wmma_bf16(a, bfr[nt], acc[nt]);
  }

  // epilogue: bias add, cast, store [B,H,S,A]
  const int rbase = half * 8;
  #pragma unroll
  for (int nt = 0; nt < 4; ++nt) {
    const int n  = nt * 16 + row16;
    const float bi = bias[h * A_ + n];
    #pragma unroll
    for (int r = 0; r < 8; ++r) {
      const int s = m0 + w * 16 + rbase + r;
      out[(((size_t)b * H_ + h) * S_ + s) * A_ + n] = (__bf16)(acc[nt][r] + bi);
    }
  }
}

// ---------------------------------------------------------------------------
// Flash attention: per (b,h), 128 query rows per WG, 8 waves x 16-row strips,
// streaming 32-key tiles, double-buffered V staging, online softmax.
// ---------------------------------------------------------------------------
__global__ __launch_bounds__(256) void k_attn(
    const __bf16* __restrict__ qb, const __bf16* __restrict__ kbuf,
    const __bf16* __restrict__ vbuf,           // [B,H,S,A]
    __bf16* __restrict__ zb)                   // [B,S,H*A]
{
  __shared__ __bf16 Vt[2][64 * 40];   // V tile transposed [A=64][32 keys]
  __shared__ __bf16 Pt[8][16 * 40];   // per-wave P staging [16 rows][32 keys]

  const int mt   = blockIdx.x;
  const int b    = blockIdx.y >> 4;
  const int h    = blockIdx.y & 15;
  const int tid  = threadIdx.x;
  const int lane = tid & 31;
  const int w    = tid >> 5;
  const int m0   = mt * 128;
  const int strip = m0 + w * 16;

  const int row16 = lane & 15;
  const int half  = lane >> 4;
  const int rbase = half * 8;

  const size_t bh = ((size_t)b * H_ + h) * (size_t)S_ * A_;
  const __bf16* Q = qb   + bh;
  const __bf16* K = kbuf + bh;
  const __bf16* V = vbuf + bh;

  // preload Q fragments (A=64 -> two K-chunks of 32) straight from global
  Frag16 qf[2];
  {
    const __bf16* qrow = Q + (size_t)(strip + row16) * A_;
    #pragma unroll
    for (int c = 0; c < 2; ++c) {
      qf[c].q[0] = *(const uint4*)(qrow + c * 32 + half * 16);
      qf[c].q[1] = *(const uint4*)(qrow + c * 32 + half * 16 + 8);
    }
  }

  float mrun[8], lrun[8];
  #pragma unroll
  for (int r = 0; r < 8; ++r) { mrun[r] = -3.0e38f; lrun[r] = 0.f; }
  f32x8 accz[4] = { zero8(), zero8(), zero8(), zero8() };

  const int vk = tid & 31;          // key within tile
  const int va = (tid >> 5) * 8;    // A-group

  auto stageV = [&](int buf, int t0) {
    Pack8 p;
    p.q = *(const uint4*)(V + (size_t)(t0 + vk) * A_ + va);
    #pragma unroll
    for (int j = 0; j < 8; ++j)
      Vt[buf][(va + j) * 40 + vk] = p.e[j];
  };

  const int nT = (m0 + 128) / 32;   // uniform across the workgroup
  stageV(0, 0);

  for (int it = 0; it < nT; ++it) {
    const int t0  = it * 32;
    const int cur = it & 1;
    __syncthreads();
    if (it + 1 < nT) {
      stageV(cur ^ 1, t0 + 32);
      if (tid < 32) __builtin_prefetch(K + (size_t)(t0 + 32 + tid) * A_);
    }

    // scores S = Q @ K^T (16x32 per wave); batch the 4 K-fragments
    Frag16 kf[2][2];
    #pragma unroll
    for (int c = 0; c < 2; ++c)
      #pragma unroll
      for (int nt = 0; nt < 2; ++nt) {
        const __bf16* kp = K + (size_t)(t0 + nt * 16 + row16) * A_ + c * 32 + half * 16;
        kf[c][nt].q[0] = *(const uint4*)(kp);
        kf[c][nt].q[1] = *(const uint4*)(kp + 8);
      }
    f32x8 accs[2] = { zero8(), zero8() };
    #pragma unroll
    for (int c = 0; c < 2; ++c)
      #pragma unroll
      for (int nt = 0; nt < 2; ++nt)
        accs[nt] = wmma_bf16(qf[c], kf[c][nt], accs[nt]);

    // online softmax (scale 1/sqrt(64)=0.125, causal mask = -1e5)
    float pv[2][8];
    float alpha[8];
    #pragma unroll
    for (int r = 0; r < 8; ++r) {
      const int sq = strip + rbase + r;
      float s0 = (t0 + row16      <= sq) ? accs[0][r] * 0.125f : -100000.0f;
      float s1 = (t0 + 16 + row16 <= sq) ? accs[1][r] * 0.125f : -100000.0f;
      float tm = fmaxf(s0, s1);
      #pragma unroll
      for (int off = 1; off < 16; off <<= 1)
        tm = fmaxf(tm, __shfl_xor(tm, off, 32));
      const float nm = fmaxf(mrun[r], tm);
      alpha[r] = __expf(mrun[r] - nm);
      const float p0 = __expf(s0 - nm);
      const float p1 = __expf(s1 - nm);
      float ts = p0 + p1;
      #pragma unroll
      for (int off = 1; off < 16; off <<= 1)
        ts += __shfl_xor(ts, off, 32);
      lrun[r] = lrun[r] * alpha[r] + ts;
      mrun[r] = nm;
      pv[0][r] = p0;
      pv[1][r] = p1;
    }
    #pragma unroll
    for (int a4 = 0; a4 < 4; ++a4)
      #pragma unroll
      for (int r = 0; r < 8; ++r)
        accz[a4][r] *= alpha[r];

    // C-layout -> A-fragment reshape of P via per-wave LDS
    __bf16* pt = &Pt[w][0];
    #pragma unroll
    for (int nt = 0; nt < 2; ++nt)
      #pragma unroll
      for (int r = 0; r < 8; ++r)
        pt[(rbase + r) * 40 + nt * 16 + row16] = (__bf16)pv[nt][r];
    wait_ds0();  // intra-wave cross-lane LDS reuse; wave32 lockstep

    Frag16 pf;
    pf.q[0] = *(const uint4*)(pt + row16 * 40 + half * 16);
    pf.q[1] = *(const uint4*)(pt + row16 * 40 + half * 16 + 8);

    // z += P @ V ; batch the 4 V-fragments
    Frag16 vf[4];
    #pragma unroll
    for (int a4 = 0; a4 < 4; ++a4) {
      const __bf16* vp = &Vt[cur][(a4 * 16 + row16) * 40 + half * 16];
      vf[a4].q[0] = *(const uint4*)(vp);
      vf[a4].q[1] = *(const uint4*)(vp + 8);
    }
    #pragma unroll
    for (int a4 = 0; a4 < 4; ++a4)
      accz[a4] = wmma_bf16(pf, vf[a4], accz[a4]);
  }

  // finalize: divide by softmax sum, store z[b,s,h*64+a]
  #pragma unroll
  for (int r = 0; r < 8; ++r) {
    const float inv = 1.0f / lrun[r];
    const int s = strip + rbase + r;
    #pragma unroll
    for (int a4 = 0; a4 < 4; ++a4)
      zb[((size_t)b * S_ + s) * (H_ * A_) + h * A_ + a4 * 16 + row16] =
          (__bf16)(accz[a4][r] * inv);
  }
}

// ---------------------------------------------------------------------------
// Output projection: out[m, d] = z[m, :] @ Wo_flat[:, d] + bo[d]  (fp32 out)
// grid: (B*S/128, D/64), block 256. Double-buffered, async A staging.
// ---------------------------------------------------------------------------
__global__ __launch_bounds__(256) void k_oproj(
    const __bf16* __restrict__ zb,   // [B*S, 1024]
    const __bf16* __restrict__ wo,   // [1024, 1024] (h*A+a major, d minor)
    const float* __restrict__ bo,    // [1024]
    float* __restrict__ out)         // [B*S, 1024]
{
  __shared__ __bf16 At[2][128 * 40];
  __shared__ __bf16 Bt[2][64 * 40];

  const int mt  = blockIdx.x;
  const int nt0 = blockIdx.y;
  const int tid  = threadIdx.x;
  const int lane = tid & 31;
  const int w    = tid >> 5;
  const int m0 = mt * 128;
  const int n0 = nt0 * 64;

  f32x8 acc[4] = { zero8(), zero8(), zero8(), zero8() };

  const int ar  = tid >> 1;
  const int ah  = tid & 1;
  const int bn  = tid & 63;
  const int bk8 = (tid >> 6) * 8;
  const int row16 = lane & 15;
  const int half  = lane >> 4;

  auto stageA = [&](int buf, int kk) {
    const __bf16* src = zb + (size_t)(m0 + ar) * (H_ * A_) + kk + ah * 16;
    __bf16* dst = &At[buf][ar * 40 + ah * 16];
    async_b128(dst, src);
    async_b128(dst + 8, src + 8);
  };
  auto stageB = [&](int buf, int kk) {
    #pragma unroll
    for (int j = 0; j < 8; ++j)
      Bt[buf][bn * 40 + bk8 + j] = wo[(size_t)(kk + bk8 + j) * D_ + n0 + bn];
  };

  stageA(0, 0);
  stageB(0, 0);

  for (int it = 0; it < (H_ * A_) / 32; ++it) {
    const int kk  = it * 32;
    const int cur = it & 1;
    wait_async0();
    __syncthreads();
    if (kk + 32 < (H_ * A_)) {
      stageA(cur ^ 1, kk + 32);
      stageB(cur ^ 1, kk + 32);
      if (kk + 64 < (H_ * A_)) {
        __builtin_prefetch(zb + (size_t)(m0 + ar) * (H_ * A_) + kk + 64 + ah * 16);
        if (tid < 32) __builtin_prefetch(wo + (size_t)(kk + 64 + tid) * D_ + n0);
      }
    }

    Frag16 a;
    {
      const __bf16* p = &At[cur][(w * 16 + row16) * 40 + half * 16];
      a.q[0] = *(const uint4*)(p);
      a.q[1] = *(const uint4*)(p + 8);
    }
    Frag16 bfr[4];
    #pragma unroll
    for (int nt = 0; nt < 4; ++nt) {
      const __bf16* p = &Bt[cur][(nt * 16 + row16) * 40 + half * 16];
      bfr[nt].q[0] = *(const uint4*)(p);
      bfr[nt].q[1] = *(const uint4*)(p + 8);
    }
    #pragma unroll
    for (int nt = 0; nt < 4; ++nt)
      acc[nt] = wmma_bf16(a, bfr[nt], acc[nt]);
  }

  const int rbase = half * 8;
  #pragma unroll
  for (int nt = 0; nt < 4; ++nt) {
    const int n = n0 + nt * 16 + row16;
    const float bi = bo[n];
    #pragma unroll
    for (int r = 0; r < 8; ++r) {
      const int m = m0 + w * 16 + rbase + r;
      out[(size_t)m * D_ + n] = acc[nt][r] + bi;
    }
  }
}

// ---------------------------------------------------------------------------
// host launcher
// ---------------------------------------------------------------------------
extern "C" void kernel_launch(void* const* d_in, const int* in_sizes, int n_in,
                              void* d_out, int out_size, void* d_ws, size_t ws_size,
                              hipStream_t stream) {
  (void)in_sizes; (void)n_in; (void)out_size; (void)ws_size;

  const float* residual = (const float*)d_in[0];   // [B,S,D]
  const float* Wq = (const float*)d_in[1];         // [H,D,A]
  const float* Wk = (const float*)d_in[2];
  const float* Wv = (const float*)d_in[3];
  const float* Wo = (const float*)d_in[4];         // [H,A,D]
  const float* bq = (const float*)d_in[5];         // [H,A]
  const float* bk = (const float*)d_in[6];
  const float* bv = (const float*)d_in[7];
  const float* bo = (const float*)d_in[8];         // [D]
  float* out = (float*)d_out;

  constexpr size_t nRes = (size_t)B_ * S_ * D_;      // 4194304
  constexpr size_t nW   = (size_t)H_ * D_ * A_;      // 1048576
  constexpr size_t nQKV = (size_t)B_ * H_ * S_ * A_; // 4194304
  constexpr size_t nZ   = (size_t)B_ * S_ * H_ * A_; // 4194304

  char* ws = (char*)d_ws;
  size_t off = 0;
  __bf16* xb  = (__bf16*)(ws + off); off += nRes * 2;
  __bf16* wqb = (__bf16*)(ws + off); off += nW * 2;
  __bf16* wkb = (__bf16*)(ws + off); off += nW * 2;
  __bf16* wvb = (__bf16*)(ws + off); off += nW * 2;
  __bf16* wob = (__bf16*)(ws + off); off += nW * 2;
  __bf16* qbf = (__bf16*)(ws + off); off += nQKV * 2;
  __bf16* kbf = (__bf16*)(ws + off); off += nQKV * 2;
  __bf16* vbf = (__bf16*)(ws + off); off += nQKV * 2;
  __bf16* zbf = (__bf16*)(ws + off); off += nZ * 2;

  // 1) fp32 -> bf16 conversions
  k_cvt<<<(int)(nRes / 4 / 256), 256, 0, stream>>>(residual, xb, (int)(nRes / 4));
  k_cvt<<<(int)(nW / 4 / 256), 256, 0, stream>>>(Wq, wqb, (int)(nW / 4));
  k_cvt<<<(int)(nW / 4 / 256), 256, 0, stream>>>(Wk, wkb, (int)(nW / 4));
  k_cvt<<<(int)(nW / 4 / 256), 256, 0, stream>>>(Wv, wvb, (int)(nW / 4));
  k_cvt<<<(int)(nW / 4 / 256), 256, 0, stream>>>(Wo, wob, (int)(nW / 4));

  // 2) QKV projection
  dim3 gq(B_ * (S_ / 128), H_, 3);
  k_qkv<<<gq, 256, 0, stream>>>(xb, wqb, wkb, wvb, bq, bk, bv, qbf, kbf, vbf);

  // 3) flash attention
  dim3 ga(S_ / 128, B_ * H_);
  k_attn<<<ga, 256, 0, stream>>>(qbf, kbf, vbf, zbf);

  // 4) output projection + bias
  dim3 go(B_ * S_ / 128, D_ / 64);
  k_oproj<<<go, 256, 0, stream>>>(zbf, wob, bo, out);
}
